// Fixformer_44959717654748
// MI455X (gfx1250) — matrix operand order repaired and checked by reading, not compiled, not run on hardware
//
#include <hip/hip_runtime.h>
#include <hip/hip_bf16.h>

typedef __attribute__((ext_vector_type(16))) _Float16 v16h;
typedef __attribute__((ext_vector_type(8)))  float    v8f;

#define BATCH 2
#define HH 256
#define WW 256
#define CH 192
#define HEADS 6
#define WIN 8
#define SHIFT 4
#define HID 768
#define NTOK 64
#define HD 32
#define MTOK (BATCH * HH * WW)      // 131072
#define SCALE 0.17677669529663689f  // 32^-0.5

// ---------------------------------------------------------------------------
// WMMA helpers
// ---------------------------------------------------------------------------
__device__ __forceinline__ v8f wmma32(v16h a, v16h b, v8f c) {
  return __builtin_amdgcn_wmma_f32_16x16x32_f16(false, a, false, b, (short)0, c,
                                                false, false);
}

// A fragment: 16x32 f16 tile, row-major, row stride ldk (even) elements.
// lane L (0..31): row = L&15, K-half = L>>4. VGPR j<4: K = half*8 + 2j;
// j>=4: K = 16 + half*8 + 2(j-4).  (ISA 7.12.2, 16-bit A 16x32)
__device__ __forceinline__ v16h frag_a(const _Float16* __restrict__ p, int ldk) {
  int lane = threadIdx.x & 31;
  int row  = lane & 15;
  int khu  = (lane >> 4) * 4;  // kh/2 in uints
  union { v16h h; unsigned u[8]; } f;
  const unsigned* q = (const unsigned*)(p + (size_t)row * ldk);
#pragma unroll
  for (int j = 0; j < 4; ++j) {
    f.u[j]     = q[khu + j];
    f.u[4 + j] = q[8 + khu + j];
  }
  return f.h;
}

// B fragment from K-contiguous storage: tile stored as [N][K] (k contiguous),
// row stride ldk (even). lane L: col = L&15, half = L>>4. VGPR j<4: K = half*8 + 2j; j>=4: K = 16 + half*8 + 2(j-4).
__device__ __forceinline__ v16h frag_bt(const _Float16* __restrict__ p, int ldk) {
  int lane = threadIdx.x & 31;
  int col  = lane & 15;
  int kbu  = (lane >> 4) * 4;  // kh/2 in uints
  union { v16h h; unsigned u[8]; } f;
  const unsigned* q = (const unsigned*)(p + (size_t)col * ldk);
#pragma unroll
  for (int j = 0; j < 4; ++j) {
    f.u[j]     = q[kbu + j];
    f.u[4 + j] = q[8 + kbu + j];
  }
  return f.h;
}

__device__ __forceinline__ float gelu_f(float x) {
  return 0.5f * x * (1.0f + erff(x * 0.70710678118654752f));
}

// ---------------------------------------------------------------------------
// Weight packing / conversion
// ---------------------------------------------------------------------------
__global__ void pack_qkv_w(const float* __restrict__ qw, const float* __restrict__ qb,
                           const float* __restrict__ kvw, const float* __restrict__ kvb,
                           _Float16* __restrict__ w, float* __restrict__ bias) {
  int idx = blockIdx.x * 256 + threadIdx.x;
  if (idx < CH * 576) {
    int k = idx / 576, j = idx % 576;
    float v = (j < CH) ? qw[k * CH + j] : kvw[k * 384 + (j - CH)];
    w[idx] = (_Float16)v;
  }
  if (idx < 576) bias[idx] = (idx < CH) ? qb[idx] : kvb[idx - CH];
}

__global__ void cvt_f16(const float* __restrict__ in, _Float16* __restrict__ out, int n) {
  int idx = blockIdx.x * 256 + threadIdx.x;
  if (idx < n) out[idx] = (_Float16)in[idx];
}

// ---------------------------------------------------------------------------
// LayerNorm over C=192: one wave per token, 6 channels per lane.
// ---------------------------------------------------------------------------
__global__ __launch_bounds__(256) void ln_kernel(const float* __restrict__ x,
                                                 const float* __restrict__ w,
                                                 const float* __restrict__ b,
                                                 _Float16* __restrict__ out) {
  int wave = threadIdx.x >> 5, lane = threadIdx.x & 31;
  size_t token = (size_t)blockIdx.x * 8 + wave;
  const float* p = x + token * CH;
  float v[6];
  float s = 0.f;
#pragma unroll
  for (int i = 0; i < 6; ++i) { v[i] = p[lane + 32 * i]; s += v[i]; }
#pragma unroll
  for (int m = 16; m >= 1; m >>= 1) s += __shfl_xor(s, m, 32);
  float mean = s * (1.0f / CH);
  float var = 0.f;
#pragma unroll
  for (int i = 0; i < 6; ++i) { float d = v[i] - mean; var += d * d; }
#pragma unroll
  for (int m = 16; m >= 1; m >>= 1) var += __shfl_xor(var, m, 32);
  float inv = rsqrtf(var * (1.0f / CH) + 1e-5f);
  _Float16* o = out + token * CH;
#pragma unroll
  for (int i = 0; i < 6; ++i) {
    int c = lane + 32 * i;
    o[c] = (_Float16)((v[i] - mean) * inv * w[c] + b[c]);
  }
}

// ---------------------------------------------------------------------------
// Generic WMMA GEMM: out[M,N] = epi(A[M,K]*B[K,N] + bias).
// FLAGS bit0: f16 output; bit1: add f32 residual (f32 out); bit2: GELU.
// Block: 256 threads, 128x64 tile, 8 waves in 4(M)x2(N), each wave 32x32
// -> 4 WMMAs per K-step with A/B fragment reuse.
// ---------------------------------------------------------------------------
template <int FLAGS>
__global__ __launch_bounds__(256) void gemm_kernel(
    const _Float16* __restrict__ A, const _Float16* __restrict__ Bw,
    const float* __restrict__ bias, float* __restrict__ outf,
    _Float16* __restrict__ outh, const float* __restrict__ resid,
    int N, int K) {
  __shared__ _Float16 As[128][32];      // 8 KB
  __shared__ _Float16 Bs[64][34];       // transposed [n][k], padded (even) 4.25 KB
  const int bm = blockIdx.x * 128;
  const int bn = blockIdx.y * 64;
  const int tid = threadIdx.x;
  const int wave = tid >> 5, lane = tid & 31;
  const int wm = wave & 3, wn = wave >> 2;

  v8f acc00 = {}, acc01 = {}, acc10 = {}, acc11 = {};

  const int ar = tid >> 2, asg = (tid & 3) * 8;    // A: row (0..63), 8-f16 seg
  const int bkr = tid >> 3, bsg = (tid & 7) * 8;   // B: k-row, 8-f16 segment

  for (int k0 = 0; k0 < K; k0 += 32) {
    // stage A tile: two rows per thread (coalesced uint4)
    const uint4 av0 = *(const uint4*)(A + (size_t)(bm + ar) * K + k0 + asg);
    const uint4 av1 = *(const uint4*)(A + (size_t)(bm + 64 + ar) * K + k0 + asg);
    *(uint4*)&As[ar][asg] = av0;
    *(uint4*)&As[64 + ar][asg] = av1;
    // stage B tile transposed
    uint4 bv = *(const uint4*)(Bw + (size_t)(k0 + bkr) * N + bn + bsg);
    const _Float16* bh = (const _Float16*)&bv;
#pragma unroll
    for (int e = 0; e < 8; ++e) Bs[bsg + e][bkr] = bh[e];
    if (k0 + 32 < K) {  // hint next tiles into cache (global_prefetch_b8)
      __builtin_prefetch(A + (size_t)(bm + ar) * K + k0 + 32 + asg, 0, 0);
      __builtin_prefetch(A + (size_t)(bm + 64 + ar) * K + k0 + 32 + asg, 0, 0);
      __builtin_prefetch(Bw + (size_t)(k0 + 32 + bkr) * N + bn + bsg, 0, 0);
    }
    __syncthreads();
    v16h a0 = frag_a(&As[wm * 32][0], 32);
    v16h a1 = frag_a(&As[wm * 32 + 16][0], 32);
    v16h b0 = frag_bt(&Bs[wn * 32][0], 34);
    v16h b1 = frag_bt(&Bs[wn * 32 + 16][0], 34);
    acc00 = wmma32(a0, b0, acc00);
    acc01 = wmma32(a0, b1, acc01);
    acc10 = wmma32(a1, b0, acc10);
    acc11 = wmma32(a1, b1, acc11);
    __syncthreads();
  }

  // epilogue: lane = N col within 16, VGPR r = M row (r + 8*(lane>=16))
  const int c0 = lane & 15;
  const int rtop = bm + wm * 32 + ((lane >> 4) << 3);
#pragma unroll
  for (int sm = 0; sm < 2; ++sm) {
#pragma unroll
    for (int sn = 0; sn < 2; ++sn) {
      v8f acc = sm ? (sn ? acc11 : acc10) : (sn ? acc01 : acc00);
      int col = bn + wn * 32 + sn * 16 + c0;
      float bv = bias[col];
      int rbase = rtop + sm * 16;
#pragma unroll
      for (int r = 0; r < 8; ++r) {
        size_t row = rbase + r;
        float v = acc[r] + bv;
        if (FLAGS & 4) v = gelu_f(v);
        if (FLAGS & 2) v += resid[row * N + col];
        if (FLAGS & 1) outh[row * N + col] = (_Float16)v;
        else           outf[row * N + col] = v;
      }
    }
  }
}

// ---------------------------------------------------------------------------
// Shifted-window attention: one block per window, 6 waves = 6 heads.
// qkv layout per token: [0,192)=Q, [192,384)=K, [384,576)=V; ch = head*32+d.
// V is staged transposed (svt[ch][token]) so P@V B-fragments are contiguous.
// ---------------------------------------------------------------------------
__global__ __launch_bounds__(192) void attn_kernel(
    const _Float16* __restrict__ qkv, const float* __restrict__ rpb,
    _Float16* __restrict__ y) {
  __shared__ _Float16 sqk[NTOK][384];         // Q|K, 48 KB
  __shared__ _Float16 svt[CH][66];            // V transposed, 25.3 KB
  __shared__ _Float16 sprob[HEADS][NTOK][66]; // 50.6 KB (padded, even stride)
  __shared__ float srpb[225 * HEADS];         // 5.4 KB

  const int widx = blockIdx.x;              // b*1024 + wh*32 + ww
  const int b = widx >> 10, wh = (widx >> 5) & 31, ww = widx & 31;
  const int tid = threadIdx.x;
  const int head = tid >> 5, lane = tid & 31;

  for (int i = tid; i < 225 * HEADS; i += 192) srpb[i] = rpb[i];

  // cooperative gather of shifted window: 3 threads per token, 192 f16 each.
  // parts 0/1 -> Q/K rows; part 2 -> V written transposed.
  {
    int n = tid / 3, part = tid % 3;
    int i = n >> 3, j = n & 7;
    int ah = (wh * WIN + i + SHIFT) & 255;
    int aw = (ww * WIN + j + SHIFT) & 255;
    size_t tok = ((size_t)b << 16) + (size_t)ah * 256 + aw;
    const uint4* g = (const uint4*)(qkv + tok * 576 + part * 192);
    if (part < 2) {
      uint4* s = (uint4*)&sqk[n][part * 192];
#pragma unroll
      for (int e = 0; e < 24; ++e) s[e] = g[e];
    } else {
#pragma unroll
      for (int e = 0; e < 24; ++e) {
        uint4 v = g[e];
        const _Float16* hv = (const _Float16*)&v;
#pragma unroll
        for (int q = 0; q < 8; ++q) svt[e * 8 + q][n] = hv[q];
      }
    }
  }
  __syncthreads();

  // region id for the (unshifted) Swin mask
  auto region = [&](int n) {
    int hh = wh * WIN + (n >> 3);
    int wc = ww * WIN + (n & 7);
    int rh = hh < 248 ? 0 : (hh < 252 ? 1 : 2);
    int rw = wc < 248 ? 0 : (wc < 252 ? 1 : 2);
    return rh * 3 + rw;
  };

  const int c0 = lane & 15;

  // ---- scores + softmax, one 16-row band (tM) at a time ----
  for (int tM = 0; tM < 4; ++tM) {
    v16h qf = frag_a(&sqk[tM * 16][head * HD], 384);
    v8f acc[4];
#pragma unroll
    for (int tN = 0; tN < 4; ++tN) {
      v16h kf = frag_bt(&sqk[tN * 16][192 + head * HD], 384);
      v8f z = {};
      acc[tN] = wmma32(qf, kf, z);
    }
    const int rb = tM * 16 + ((lane >> 4) << 3);
#pragma unroll
    for (int r = 0; r < 8; ++r) {
      int row = rb + r;
      int i0 = row >> 3, j0 = row & 7;
      int regR = region(row);
      float vals[4], mx = -1e30f;
#pragma unroll
      for (int tN = 0; tN < 4; ++tN) {
        int col = tN * 16 + c0;
        int rpi = (i0 - (col >> 3) + 7) * 15 + (j0 - (col & 7) + 7);
        float mk = (region(col) != regR) ? -100.f : 0.f;
        vals[tN] = acc[tN][r] * SCALE + srpb[rpi * HEADS + head] + mk;
        mx = fmaxf(mx, vals[tN]);
      }
#pragma unroll
      for (int m = 8; m >= 1; m >>= 1) mx = fmaxf(mx, __shfl_xor(mx, m, 32));
      float sum = 0.f;
#pragma unroll
      for (int tN = 0; tN < 4; ++tN) { vals[tN] = __expf(vals[tN] - mx); sum += vals[tN]; }
#pragma unroll
      for (int m = 8; m >= 1; m >>= 1) sum += __shfl_xor(sum, m, 32);
      float inv = 1.0f / sum;
#pragma unroll
      for (int tN = 0; tN < 4; ++tN)
        sprob[head][row][tN * 16 + c0] = (_Float16)(vals[tN] * inv);
    }
  }

  // ---- y = P @ V (V fragments now contiguous-in-K via svt) ----
  for (int tM = 0; tM < 4; ++tM) {
    v8f o0 = {}, o1 = {};
#pragma unroll
    for (int ks = 0; ks < 2; ++ks) {
      v16h pf = frag_a(&sprob[head][tM * 16][ks * 32], 66);
      v16h v0 = frag_bt(&svt[head * HD][ks * 32], 66);
      v16h v1 = frag_bt(&svt[head * HD + 16][ks * 32], 66);
      o0 = wmma32(pf, v0, o0);
      o1 = wmma32(pf, v1, o1);
    }
    const int rb = tM * 16 + ((lane >> 4) << 3);
#pragma unroll
    for (int r = 0; r < 8; ++r) {
      int n = rb + r;
      int ah = (wh * WIN + (n >> 3) + SHIFT) & 255;
      int aw = (ww * WIN + (n & 7) + SHIFT) & 255;
      size_t tok = ((size_t)b << 16) + (size_t)ah * 256 + aw;
      y[tok * CH + head * HD + c0]      = (_Float16)o0[r];
      y[tok * CH + head * HD + 16 + c0] = (_Float16)o1[r];
    }
  }
}

// ---------------------------------------------------------------------------
// Depthwise 3x3 conv (SAME, zero pad) + exact GELU. Two channels per thread.
// ---------------------------------------------------------------------------
__global__ __launch_bounds__(256) void dwconv_kernel(
    const _Float16* __restrict__ h, const float* __restrict__ dw,
    const float* __restrict__ db, _Float16* __restrict__ out) {
  size_t idx = (size_t)blockIdx.x * 256 + threadIdx.x;  // MTOK * 384
  int cp = (int)(idx % 384);
  size_t pix = idx / 384;
  int x = pix & 255, yy = (int)((pix >> 8) & 255), bb = (int)(pix >> 16);
  int c = cp * 2;
  float a0 = db[c], a1 = db[c + 1];
#pragma unroll
  for (int dy = -1; dy <= 1; ++dy) {
    int ys = yy + dy;
    if (ys < 0 || ys > 255) continue;
#pragma unroll
    for (int dx = -1; dx <= 1; ++dx) {
      int xs = x + dx;
      if (xs < 0 || xs > 255) continue;
      size_t off = (((size_t)bb << 16) + (size_t)ys * 256 + xs) * HID + c;
      unsigned uv = *(const unsigned*)(h + off);
      const _Float16* hv = (const _Float16*)&uv;
      int wi = ((dy + 1) * 3 + (dx + 1)) * HID + c;
      a0 += (float)hv[0] * dw[wi];
      a1 += (float)hv[1] * dw[wi + 1];
    }
  }
  size_t o = pix * HID + c;
  out[o]     = (_Float16)gelu_f(a0);
  out[o + 1] = (_Float16)gelu_f(a1);
}

// ---------------------------------------------------------------------------
// Host orchestration
// ---------------------------------------------------------------------------
extern "C" void kernel_launch(void* const* d_in, const int* in_sizes, int n_in,
                              void* d_out, int out_size, void* d_ws, size_t ws_size,
                              hipStream_t stream) {
  const float* x       = (const float*)d_in[0];
  const float* n1w     = (const float*)d_in[1];
  const float* n1b     = (const float*)d_in[2];
  const float* q_w     = (const float*)d_in[3];
  const float* q_b     = (const float*)d_in[4];
  const float* kv_w    = (const float*)d_in[5];
  const float* kv_b    = (const float*)d_in[6];
  const float* rpb     = (const float*)d_in[7];
  const float* proj_w  = (const float*)d_in[8];
  const float* proj_b  = (const float*)d_in[9];
  const float* n2w     = (const float*)d_in[10];
  const float* n2b     = (const float*)d_in[11];
  const float* l1_w    = (const float*)d_in[12];
  const float* l1_b    = (const float*)d_in[13];
  const float* dw_w    = (const float*)d_in[14];
  const float* dw_b    = (const float*)d_in[15];
  const float* l2_w    = (const float*)d_in[16];
  const float* l2_b    = (const float*)d_in[17];
  float* out = (float*)d_out;

  char* ws = (char*)d_ws;
  // lifetime-aliased workspace layout (all offsets 16B aligned)
  float*     x1    = (float*)ws;                              // 100,663,296 B
  _Float16*  wqkv  = (_Float16*)(ws + 100663296);             //     221,184 B
  float*     bqkv  = (float*)(ws + 100884480);                //       2,304 B
  _Float16*  wproj = (_Float16*)(ws + 100886784);             //      73,728 B
  _Float16*  wl1   = (_Float16*)(ws + 100960512);             //     294,912 B
  _Float16*  wl2   = (_Float16*)(ws + 101255424);             //     294,912 B
  char*      bufA  = ws + 101550336;                          // 201,326,592 B: xn1 -> h
  char*      bufB  = ws + 302876928;                          // 201,326,592 B: qkv -> hh
  char*      bufC  = ws + 504203520;                          //  50,331,648 B: y -> xn2

  _Float16* xn1  = (_Float16*)bufA;
  _Float16* hbuf = (_Float16*)bufA;
  _Float16* qkvb = (_Float16*)bufB;
  _Float16* hhb  = (_Float16*)bufB;
  _Float16* ybuf = (_Float16*)bufC;
  _Float16* xn2  = (_Float16*)bufC;

  // 1) weight conversion
  pack_qkv_w<<<(CH * 576 + 255) / 256, 256, 0, stream>>>(q_w, q_b, kv_w, kv_b, wqkv, bqkv);
  cvt_f16<<<(CH * CH + 255) / 256, 256, 0, stream>>>(proj_w, wproj, CH * CH);
  cvt_f16<<<(CH * HID + 255) / 256, 256, 0, stream>>>(l1_w, wl1, CH * HID);
  cvt_f16<<<(HID * CH + 255) / 256, 256, 0, stream>>>(l2_w, wl2, HID * CH);

  // 2) LN1
  ln_kernel<<<MTOK / 8, 256, 0, stream>>>(x, n1w, n1b, xn1);

  // 3) fused QKV GEMM: [M,192] x [192,576] -> f16
  gemm_kernel<1><<<dim3(MTOK / 128, 576 / 64), 256, 0, stream>>>(
      xn1, wqkv, bqkv, nullptr, qkvb, nullptr, 576, CH);

  // 4) shifted-window attention (2048 windows)
  attn_kernel<<<BATCH * 32 * 32, 192, 0, stream>>>(qkvb, rpb, ybuf);

  // 5) proj GEMM + residual: x1 = x + y @ proj_w + b
  gemm_kernel<2><<<dim3(MTOK / 128, CH / 64), 256, 0, stream>>>(
      ybuf, wproj, proj_b, x1, nullptr, x, CH, CH);

  // 6) LN2
  ln_kernel<<<MTOK / 8, 256, 0, stream>>>(x1, n2w, n2b, xn2);

  // 7) l1 GEMM + GELU -> h f16 [M,768]
  gemm_kernel<5><<<dim3(MTOK / 128, HID / 64), 256, 0, stream>>>(
      xn2, wl1, l1_b, nullptr, hbuf, nullptr, HID, CH);

  // 8) depthwise 3x3 + GELU -> hh f16
  dwconv_kernel<<<(unsigned)((size_t)MTOK * 384 / 256), 256, 0, stream>>>(
      hbuf, dw_w, dw_b, hhb);

  // 9) l2 GEMM + residual -> out f32
  gemm_kernel<2><<<dim3(MTOK / 128, CH / 64), 256, 0, stream>>>(
      hhb, wl2, l2_b, out, nullptr, x1, CH, HID);
}
